// GATLayer_63299228009033
// MI455X (gfx1250) — matrix operand order repaired and checked by reading, not compiled, not run on hardware
//
#include <hip/hip_runtime.h>

typedef __attribute__((ext_vector_type(16))) __bf16 v16bf;
typedef __attribute__((ext_vector_type(8)))  __bf16 v8bf;
typedef __attribute__((ext_vector_type(8)))  float  v8f;
typedef __attribute__((ext_vector_type(4)))  int    i4;
typedef __attribute__((ext_vector_type(4)))  float  f4;

#define NVAL 8192
#define DOUT 128
#define NEG_SLOPE 0.2f
#define NEG_BIG  -3.0e38f
#define LOG2E 1.4426950408889634f

// ---------------------------------------------------------------------------
// Kernel 0: coop_norm[i] = sigmoid(w * coop_loss[i] + b)
// ---------------------------------------------------------------------------
__global__ void coop_kernel(const float* __restrict__ loss,
                            const float* __restrict__ w,
                            const float* __restrict__ b,
                            float* __restrict__ coop_norm) {
  int i = blockIdx.x * blockDim.x + threadIdx.x;
  if (i < NVAL) {
    float x = w[0] * loss[i] + b[0];
    coop_norm[i] = 1.0f / (1.0f + __expf(-x));
  }
}

// ---------------------------------------------------------------------------
// Kernel T: Wt[n][k] = bf16(W[k][n])   (W stored (in,out) row-major)
// ---------------------------------------------------------------------------
__global__ void transpose_w_kernel(const float* __restrict__ W,
                                   unsigned short* __restrict__ Wt_raw,
                                   int d_in) {
  __bf16* Wt = (__bf16*)Wt_raw;
  int idx = blockIdx.x * blockDim.x + threadIdx.x;
  if (idx < d_in * DOUT) {
    int n = idx / d_in;
    int k = idx - n * d_in;
    Wt[(size_t)n * d_in + k] = (__bf16)W[(size_t)k * DOUT + n];
  }
}

// ---------------------------------------------------------------------------
// Kernel A: per 16-row tile (one wave each):
//   Wh = X @ W            (WMMA bf16, f32 accum)
//   row_add[m] = LOG2E * (Wh[m,:]·a_src + coop_norm[m])   (log2-domain scores)
//   col_add[m] = LOG2E * (Wh[m,:]·a_dst + coop_norm[m])
//   WhT (bf16, [DOUT][NVAL]) for the flash kernel's B fragments
// ---------------------------------------------------------------------------
__global__ __launch_bounds__(128) void gemm_prep_kernel(
    const float* __restrict__ X, const unsigned short* __restrict__ Wt_raw,
    const float* __restrict__ a_src, const float* __restrict__ a_dst,
    const float* __restrict__ coop_norm,
    unsigned short* __restrict__ WhT_raw,
    float* __restrict__ row_add, float* __restrict__ col_add, int d_in) {
  const __bf16* Wt  = (const __bf16*)Wt_raw;
  __bf16*       WhT = (__bf16*)WhT_raw;

  const int lane = threadIdx.x & 31;
  const int wave = (blockIdx.x * blockDim.x + threadIdx.x) >> 5;
  const int row0 = wave * 16;
  const int half = lane >> 4;    // which 16-lane group
  const int lm   = lane & 15;    // A: row M ; B: column N

  v8f acc[8];
#pragma unroll
  for (int t = 0; t < 8; t++) acc[t] = (v8f)0.0f;

  const float* xrow = X + (size_t)(row0 + lm) * d_in;

  for (int k0 = 0; k0 < d_in; k0 += 32) {
    // A fragment: lane holds K = 16h+0..15 contiguous for row lm
    f4 xa = *(const f4*)(xrow + k0 + 16 * half);
    f4 xb = *(const f4*)(xrow + k0 + 16 * half + 4);
    f4 xc = *(const f4*)(xrow + k0 + 16 * half + 8);
    f4 xd = *(const f4*)(xrow + k0 + 16 * half + 12);
    v16bf a;
    a[0] = (__bf16)xa[0]; a[1] = (__bf16)xa[1]; a[2]  = (__bf16)xa[2]; a[3]  = (__bf16)xa[3];
    a[4] = (__bf16)xb[0]; a[5] = (__bf16)xb[1]; a[6]  = (__bf16)xb[2]; a[7]  = (__bf16)xb[3];
    a[8] = (__bf16)xc[0]; a[9] = (__bf16)xc[1]; a[10] = (__bf16)xc[2]; a[11] = (__bf16)xc[3];
    a[12]= (__bf16)xd[0]; a[13]= (__bf16)xd[1]; a[14] = (__bf16)xd[2]; a[15] = (__bf16)xd[3];
#pragma unroll
    for (int t = 0; t < 8; t++) {
      // B fragment: lane holds K = 16*half .. +15 contiguous for column lm
      v16bf b = *(const v16bf*)(Wt + (size_t)(t * 16 + lm) * d_in + k0 + 16 * half);
      acc[t] = __builtin_amdgcn_wmma_f32_16x16x32_bf16(
          false, a, false, b, (short)0, acc[t], false, false);
    }
  }

  // src/dst row dot products. D layout: lane L, elem r -> row 8*half + r, col 16t + lm.
  float sp[8], dp[8];
#pragma unroll
  for (int r = 0; r < 8; r++) { sp[r] = 0.0f; dp[r] = 0.0f; }
#pragma unroll
  for (int t = 0; t < 8; t++) {
    float as = a_src[t * 16 + lm];
    float ad = a_dst[t * 16 + lm];
#pragma unroll
    for (int r = 0; r < 8; r++) {
      sp[r] += acc[t][r] * as;
      dp[r] += acc[t][r] * ad;
    }
  }
#pragma unroll
  for (int mask = 1; mask <= 8; mask <<= 1) {
#pragma unroll
    for (int r = 0; r < 8; r++) {
      sp[r] += __shfl_xor(sp[r], mask, 32);
      dp[r] += __shfl_xor(dp[r], mask, 32);
    }
  }
  if (lm == 0) {
#pragma unroll
    for (int r = 0; r < 8; r++) {
      int m = row0 + 8 * half + r;
      float cn = coop_norm[m];
      row_add[m] = (sp[r] + cn) * LOG2E;   // pre-scaled: inner loop uses exp2
      col_add[m] = (dp[r] + cn) * LOG2E;
    }
  }

  // WhT bf16 store: column n = 16t+lm, 8 consecutive rows per lane -> one b128
#pragma unroll
  for (int t = 0; t < 8; t++) {
    v8bf pk;
#pragma unroll
    for (int r = 0; r < 8; r++) pk[r] = (__bf16)acc[t][r];
    *(v8bf*)(WhT + (size_t)(t * 16 + lm) * NVAL + row0 + 8 * half) = pk;
  }
}

// ---------------------------------------------------------------------------
// Kernel B: fused masked-softmax attention (flash style), one wave per 16 rows
//   out = elu( softmax_row( mask(leakyrelu(row_add_i + col_add_j), adj) ) @ Wh )
// Scores live in the log2 domain (prescaled by LOG2E) -> raw v_exp_f32.
// adj is a 256MB one-shot stream -> non-temporal loads so it can't thrash L2
// (WhT must stay L2-resident: 2MB re-read 512x per layer).
// ---------------------------------------------------------------------------
__device__ __forceinline__ float gat_score(float ra, float c, int m) {
  float e = ra + c;
  e = e > 0.0f ? e : NEG_SLOPE * e;
  return (m != 0) ? e : NEG_BIG;
}

__global__ __launch_bounds__(128) void flash_gat_kernel(
    const int* __restrict__ adj, const float* __restrict__ row_add,
    const float* __restrict__ col_add, const unsigned short* __restrict__ WhT_raw,
    float* __restrict__ H) {
  const __bf16* WhT = (const __bf16*)WhT_raw;

  const int lane = threadIdx.x & 31;
  const int wave = (blockIdx.x * blockDim.x + threadIdx.x) >> 5;
  const int row0 = wave * 16;
  const int half = lane >> 4;
  const int lm   = lane & 15;

  v8f acc[8];
#pragma unroll
  for (int t = 0; t < 8; t++) acc[t] = (v8f)0.0f;

  float m_run = -1.0e30f, l_run = 0.0f;
  const float ra = row_add[row0 + lm];
  const int* adjrow = adj + (size_t)(row0 + lm) * NVAL;

  for (int j = 0; j < NVAL; j += 32) {
    const int c0 = j + 16 * half;       // A-frag K = 16h..16h+7
    const int c1 = j + 16 * half + 8;   // A-frag K = 16h+8..16h+15
    f4 ca0 = *(const f4*)(col_add + c0);
    f4 ca1 = *(const f4*)(col_add + c0 + 4);
    f4 cb0 = *(const f4*)(col_add + c1);
    f4 cb1 = *(const f4*)(col_add + c1 + 4);
    i4 m0 = __builtin_nontemporal_load((const i4*)(adjrow + c0));
    i4 m1 = __builtin_nontemporal_load((const i4*)(adjrow + c0 + 4));
    i4 m2 = __builtin_nontemporal_load((const i4*)(adjrow + c1));
    i4 m3 = __builtin_nontemporal_load((const i4*)(adjrow + c1 + 4));
    if (j + 256 < NVAL) __builtin_prefetch(adjrow + c0 + 256, 0, 0);

    float s[16];
    s[0]  = gat_score(ra, ca0[0], m0[0]); s[1]  = gat_score(ra, ca0[1], m0[1]);
    s[2]  = gat_score(ra, ca0[2], m0[2]); s[3]  = gat_score(ra, ca0[3], m0[3]);
    s[4]  = gat_score(ra, ca1[0], m1[0]); s[5]  = gat_score(ra, ca1[1], m1[1]);
    s[6]  = gat_score(ra, ca1[2], m1[2]); s[7]  = gat_score(ra, ca1[3], m1[3]);
    s[8]  = gat_score(ra, cb0[0], m2[0]); s[9]  = gat_score(ra, cb0[1], m2[1]);
    s[10] = gat_score(ra, cb0[2], m2[2]); s[11] = gat_score(ra, cb0[3], m2[3]);
    s[12] = gat_score(ra, cb1[0], m3[0]); s[13] = gat_score(ra, cb1[1], m3[1]);
    s[14] = gat_score(ra, cb1[2], m3[2]); s[15] = gat_score(ra, cb1[3], m3[3]);

    float cmax = s[0];
#pragma unroll
    for (int k = 1; k < 16; k++) cmax = fmaxf(cmax, s[k]);
    cmax = fmaxf(cmax, __shfl_xor(cmax, 16, 32));   // both halves of row
    float m_new = fmaxf(m_run, cmax);

    float corr = 1.0f;
    if (__any(m_new > m_run)) {                     // wave-uniform rescale path
      corr = __builtin_amdgcn_exp2f(m_run - m_new);
      float rs[8];
#pragma unroll
      for (int r = 0; r < 8; r++) rs[r] = __shfl(corr, 8 * half + r, 32);
#pragma unroll
      for (int t = 0; t < 8; t++)
#pragma unroll
        for (int r = 0; r < 8; r++) acc[t][r] *= rs[r];
      m_run = m_new;
    }

    float psum = 0.0f;
    v16bf a;
#pragma unroll
    for (int k = 0; k < 16; k++) {
      float p = __builtin_amdgcn_exp2f(s[k] - m_run);   // log2-domain scores
      psum += p;
      a[k] = (__bf16)p;
    }
    psum += __shfl_xor(psum, 16, 32);
    l_run = l_run * corr + psum;

#pragma unroll
    for (int t = 0; t < 8; t++) {
      // B fragment: Wh^T column block, lane holds K = j + 16*half .. +15
      v16bf b = *(const v16bf*)(WhT + (size_t)(t * 16 + lm) * NVAL + j + 16 * half);
      acc[t] = __builtin_amdgcn_wmma_f32_16x16x32_bf16(
          false, a, false, b, (short)0, acc[t], false, false);
    }
  }

  float linv = 1.0f / l_run;
  float rl[8];
#pragma unroll
  for (int r = 0; r < 8; r++) rl[r] = __shfl(linv, 8 * half + r, 32);
#pragma unroll
  for (int t = 0; t < 8; t++) {
#pragma unroll
    for (int r = 0; r < 8; r++) {
      float v = acc[t][r] * rl[r];
      v = v > 0.0f ? v : (__expf(v) - 1.0f);        // elu
      H[(size_t)(row0 + 8 * half + r) * DOUT + t * 16 + lm] = v;
    }
  }
}

// ---------------------------------------------------------------------------
extern "C" void kernel_launch(void* const* d_in, const int* in_sizes, int n_in,
                              void* d_out, int out_size, void* d_ws, size_t ws_size,
                              hipStream_t stream) {
  (void)in_sizes; (void)n_in; (void)out_size; (void)ws_size;
  const float* V         = (const float*)d_in[0];
  const int*   adj       = (const int*)d_in[1];
  const float* coop_loss = (const float*)d_in[2];
  const float* W1        = (const float*)d_in[3];
  const float* a_src1    = (const float*)d_in[4];
  const float* a_dst1    = (const float*)d_in[5];
  const float* W2        = (const float*)d_in[6];
  const float* a_src2    = (const float*)d_in[7];
  const float* a_dst2    = (const float*)d_in[8];
  const float* coop_w    = (const float*)d_in[9];
  const float* coop_b    = (const float*)d_in[10];
  float* out = (float*)d_out;

  char* ws = (char*)d_ws;
  float*          coop_norm = (float*)(ws);                      // 32 KB
  float*          row_add   = (float*)(ws + (32u << 10));        // 32 KB
  float*          col_add   = (float*)(ws + (64u << 10));        // 32 KB
  unsigned short* Wt        = (unsigned short*)(ws + (96u << 10));           // 64 KB
  unsigned short* WhT       = (unsigned short*)(ws + (160u << 10));          // 2 MB
  float*          H1        = (float*)(ws + (160u << 10) + (2048u << 10));   // 4 MB

  coop_kernel<<<NVAL / 256, 256, 0, stream>>>(coop_loss, coop_w, coop_b, coop_norm);

  // ---- Layer 1 (d_in = 256) ----
  transpose_w_kernel<<<(256 * DOUT + 255) / 256, 256, 0, stream>>>(W1, Wt, 256);
  gemm_prep_kernel<<<128, 128, 0, stream>>>(V, Wt, a_src1, a_dst1, coop_norm,
                                            WhT, row_add, col_add, 256);
  flash_gat_kernel<<<128, 128, 0, stream>>>(adj, row_add, col_add, WhT, H1);

  // ---- Layer 2 (d_in = 128) ----
  transpose_w_kernel<<<(128 * DOUT + 255) / 256, 256, 0, stream>>>(W2, Wt, 128);
  gemm_prep_kernel<<<128, 128, 0, stream>>>(H1, Wt, a_src2, a_dst2, coop_norm,
                                            WhT, row_add, col_add, 128);
  flash_gat_kernel<<<128, 128, 0, stream>>>(adj, row_add, col_add, WhT, out);
}
